// MultibandFrameAttention_5875515260934
// MI455X (gfx1250) — matrix operand order repaired and checked by reading, not compiled, not run on hardware
//
#include <hip/hip_runtime.h>
#include <cstdint>
#include <cstddef>

typedef __attribute__((ext_vector_type(16))) __bf16 v16bf;
typedef __attribute__((ext_vector_type(8)))  float  v8f;
typedef __attribute__((ext_vector_type(4)))  unsigned int v4u;
typedef __attribute__((ext_vector_type(8)))  int    v8i;
typedef __attribute__((ext_vector_type(4)))  int    v4i;

#define BATCH 4
#define SEQ   1024
#define CH    2048
#define HEADS 16
#define HDIM  128
#define MPROJ (BATCH * SEQ)   // 4096

// ---------------------------------------------------------------- helpers ---

__device__ __forceinline__ v8f zero8() {
  v8f z;
#pragma unroll
  for (int i = 0; i < 8; ++i) z[i] = 0.0f;
  return z;
}

// LDS byte offset = low 32 bits of the flat address (ISA 10.2 aperture rule).
__device__ __forceinline__ unsigned ldsaddr(const void* p) {
  return (unsigned)(unsigned long long)(uintptr_t)p;
}

// --------------------------- Tensor Data Mover: 2D tile load (ISA ch. 8) ----
// D# group0: [1:0]=count=1, [63:32]=lds_addr, [120:64]=global_addr, [127:126]=type=2
// D# group1: [17:16]=data_size(1=2B), [20]=pad_enable, [24:22]=pad_interval,
//            [31:25]=pad_amount, [79:48]=tensor_dim0, [111:80]=tensor_dim1,
//            [127:112]=tile_dim0, [143:128]=tile_dim1, [207:160]=dim0_stride
// pad_iv: 2^(c+1) DWORDs before pad; pad_amt: (c+1) DWORDs of pad.
__device__ __forceinline__ void tdm_load_2d(unsigned lds_byte_addr,
                                            const void* gptr,
                                            unsigned cols_elems, unsigned rows,
                                            unsigned row_stride_elems,
                                            unsigned pad_iv, unsigned pad_amt) {
  unsigned long long ga = (unsigned long long)(uintptr_t)gptr;
  v4u g0;
  g0[0] = 1u;                                   // count=1 (valid user D#)
  g0[1] = lds_byte_addr;
  g0[2] = (unsigned)(ga & 0xffffffffu);
  g0[3] = (unsigned)((ga >> 32) & 0x01ffffffu) | (2u << 30);   // type=2 (image)
  v8i g1;
  g1[0] = (int)((1u << 16) | (1u << 20) | (pad_iv << 22) | (pad_amt << 25));
  g1[1] = (int)((cols_elems & 0xffffu) << 16);                 // abar=0 | td0.lo
  g1[2] = (int)(((cols_elems >> 16) & 0xffffu) | ((rows & 0xffffu) << 16));
  g1[3] = (int)(((rows >> 16) & 0xffffu) | ((cols_elems & 0xffffu) << 16)); // tile_dim0
  g1[4] = (int)(rows & 0xffffu);                               // tile_dim1 | tile_dim2=0
  g1[5] = (int)row_stride_elems;                               // dim0_stride lo32
  g1[6] = 0;
  g1[7] = 0;
  v4i gz = {0, 0, 0, 0};
#if __clang_major__ >= 23
  v8i gz8 = {0, 0, 0, 0, 0, 0, 0, 0};
  __builtin_amdgcn_tensor_load_to_lds(g0, g1, gz, gz, gz8, 0);
#else
  __builtin_amdgcn_tensor_load_to_lds(g0, g1, gz, gz, 0);
#endif
}

// 16-bit A-matrix 16x32 layout (ISA 7.12.2): lane[3:0]=M, lane[4]=hh.
__device__ __forceinline__ int a_k_of_e(int e, int hh) {
  int j = e >> 1;
  int base = (j < 4) ? (j * 2) : (16 + (j - 4) * 2);
  return base + hh * 8 + (e & 1);
}

__device__ __forceinline__ v16bf load_frag_a(const __bf16* base, int m0, int k0,
                                             int sm, int sk) {
  int lane = threadIdx.x & 31;
  int m = m0 + (lane & 15);
  int hh = lane >> 4;
  v16bf r;
#pragma unroll
  for (int e = 0; e < 16; ++e)
    r[e] = base[m * sm + (k0 + a_k_of_e(e, hh)) * sk];
  return r;
}

// B-matrix 32x16: lane[3:0]=N, lanes 0-15 hold K=0..7,16..23, lanes 16-31 K=8..15,24..31.
__device__ __forceinline__ v16bf load_frag_b(const __bf16* base, int n0, int k0,
                                             int sn, int sk) {
  int lane = threadIdx.x & 31;
  int n = n0 + (lane & 15);
  int kb = k0 + (lane >> 4) * 8;
  v16bf r;
#pragma unroll
  for (int e = 0; e < 16; ++e)
    r[e] = base[n * sn + (kb + (e < 8 ? e : e + 8)) * sk];
  return r;
}

__device__ __forceinline__ v8f wmma_bf16(v16bf a, v16bf b, v8f c) {
  return __builtin_amdgcn_wmma_f32_16x16x32_bf16(false, a, false, b,
                                                 (short)0, c, false, false);
}

// ------------------------------------------------------- conversion kernels ---

__global__ void cvt_f32_bf16(const float* __restrict__ in, __bf16* __restrict__ out,
                             int n) {
  int i = blockIdx.x * blockDim.x + threadIdx.x;
  if (i < n) out[i] = (__bf16)in[i];
}

// er: [HDIM, SEQ] f32 -> ert: [SEQ, HDIM] bf16
__global__ void transpose_er(const float* __restrict__ er, __bf16* __restrict__ ert) {
  int i = blockIdx.x * blockDim.x + threadIdx.x;
  if (i < HDIM * SEQ) {
    int d = i / SEQ, j = i % SEQ;
    ert[j * HDIM + d] = (__bf16)er[i];
  }
}

// ------------------------------------------------------------ NT GEMM (WMMA) --
// C[m,n] = alpha * sum_k A[m,k]*B[n,k]; TDM-streamed, double-buffered LDS tiles.

#define GBM 128
#define GBN 128
#define GBK 32
#define GLDA 40   // LDS row stride in bf16 elems (64B data + 16B pad)

template <typename CT>
__global__ __launch_bounds__(256) void gemm_nt_kernel(
    const __bf16* __restrict__ Ag, const __bf16* __restrict__ Bg, CT* __restrict__ Cg,
    int M, int N, int K, int lda, int ldb, int ldc,
    long aO, long aI, long bO, long bI, long cO, long cI, int innerN, float alpha) {
  __shared__ __bf16 As[2][GBM * GLDA];
  __shared__ __bf16 Bs[2][GBN * GLDA];
  int zb = blockIdx.z / innerN, zi = blockIdx.z % innerN;
  const __bf16* A = Ag + zb * aO + zi * aI;
  const __bf16* B = Bg + zb * bO + zi * bI;
  CT* C = Cg + zb * cO + zi * cI;

  int tid = threadIdx.x;
  int m0 = blockIdx.y * GBM;
  int n0 = blockIdx.x * GBN;
  int wid = tid >> 5;
  int wrow = wid & 3;    // 4 wave-rows * 32 rows = 128
  int wcol = wid >> 2;   // 2 wave-cols * 64 cols = 128

  v8f acc[2][4];
#pragma unroll
  for (int i = 0; i < 2; ++i)
#pragma unroll
    for (int j = 0; j < 4; ++j) acc[i][j] = zero8();

  // TDM: tile = 32 elems/row (64B) x 128 rows; pad 16B per 64B -> GLDA stride.
  if (wid == 0) {
    tdm_load_2d(ldsaddr(&As[0][0]), A + (long)m0 * lda, GBK, GBM, lda, 3, 3);
    tdm_load_2d(ldsaddr(&Bs[0][0]), B + (long)n0 * ldb, GBK, GBN, ldb, 3, 3);
  }

  int nk = K / GBK;
  for (int it = 0; it < nk; ++it) {
    int buf = it & 1;
    if (wid == 0) {
      if (it + 1 < nk) {
        long kt = (long)(it + 1) * GBK;
        tdm_load_2d(ldsaddr(&As[buf ^ 1][0]), A + (long)m0 * lda + kt, GBK, GBM, lda, 3, 3);
        tdm_load_2d(ldsaddr(&Bs[buf ^ 1][0]), B + (long)n0 * ldb + kt, GBK, GBN, ldb, 3, 3);
        __builtin_amdgcn_s_wait_tensorcnt(2);   // current pair retired
      } else {
        __builtin_amdgcn_s_wait_tensorcnt(0);
      }
    }
    __syncthreads();
    v16bf a0 = load_frag_a(&As[buf][0], wrow * 32, 0, GLDA, 1);
    v16bf a1 = load_frag_a(&As[buf][0], wrow * 32 + 16, 0, GLDA, 1);
#pragma unroll
    for (int j = 0; j < 4; ++j) {
      v16bf bf = load_frag_b(&Bs[buf][0], wcol * 64 + j * 16, 0, GLDA, 1);
      acc[0][j] = wmma_bf16(a0, bf, acc[0][j]);
      acc[1][j] = wmma_bf16(a1, bf, acc[1][j]);
    }
    __syncthreads();   // buffer consumed before TDM refills it
  }

  int lane = tid & 31;
#pragma unroll
  for (int i = 0; i < 2; ++i)
#pragma unroll
    for (int j = 0; j < 4; ++j)
#pragma unroll
      for (int r = 0; r < 8; ++r) {
        int m = m0 + wrow * 32 + i * 16 + (lane >> 4) * 8 + r;
        int n = n0 + wcol * 64 + j * 16 + (lane & 15);
        C[(long)m * ldc + n] = (CT)(acc[i][j][r] * alpha);
      }
}

// ----------------------------------------------------- depthwise conv (k=3) ---
// grid.z selects q/k/v; bf16 in (GEMM output), bf16 out (WMMA operand).

__global__ void dwconv_bf16(const __bf16* __restrict__ yb, const float* __restrict__ w3,
                            __bf16* __restrict__ outb) {
  long toff = (long)blockIdx.z * MPROJ * CH;
  const __bf16* y = yb + toff + (long)blockIdx.y * SEQ * CH;
  __bf16* out = outb + toff + (long)blockIdx.y * SEQ * CH;
  int i = blockIdx.x * blockDim.x + threadIdx.x;  // over SEQ*CH
  int s = i / CH, ch = i % CH;
  float w0 = w3[ch * 3 + 0], w1 = w3[ch * 3 + 1], w2 = w3[ch * 3 + 2];
  float acc = w1 * (float)y[(long)s * CH + ch];
  if (s > 0)        acc += w0 * (float)y[(long)(s - 1) * CH + ch];
  if (s < SEQ - 1)  acc += w2 * (float)y[(long)(s + 1) * CH + ch];
  out[(long)s * CH + ch] = (__bf16)acc;
}

// --------------------------------------------------------- attention kernel ---
// Per (b, h, 32-row block): S[32,1024] = scale*(Q Kt + QEt) in LDS, softmax
// (1/rowsum deferred), O = P V. K/V tiles streamed by TDM, double buffered.

#define AT_R 32
#define KVLD 136   // 128 bf16 data + 8 bf16 pad = 272B rows (TDM pad 16B/256B)

#define SM_P   (32 * 1024 * 4)                 // 131072
#define SM_QS  (SM_P + 32 * 1024 * 2)          // 196608
#define SM_KV0 (SM_QS + AT_R * KVLD * 2)       // 205312
#define SM_KV1 (SM_KV0 + 128 * KVLD * 2)       // 240128
#define SM_RS  (SM_KV1 + 128 * KVLD * 2)       // 274944
#define SM_TOT (SM_RS + AT_R * 4)              // 275072

__global__ __launch_bounds__(256) void attention_kernel(
    const __bf16* __restrict__ qc, const __bf16* __restrict__ kc,
    const __bf16* __restrict__ vc, const __bf16* __restrict__ qet,
    __bf16* __restrict__ oc, float scale) {
  extern __shared__ char smem[];
  float*  S      = reinterpret_cast<float*>(smem);
  __bf16* P      = reinterpret_cast<__bf16*>(smem + SM_P);
  __bf16* Qs     = reinterpret_cast<__bf16*>(smem + SM_QS);
  __bf16* KV0p   = reinterpret_cast<__bf16*>(smem + SM_KV0);
  __bf16* KV1p   = reinterpret_cast<__bf16*>(smem + SM_KV1);
  float*  rowsum = reinterpret_cast<float*>(smem + SM_RS);

  int tid = threadIdx.x;
  int lane = tid & 31, wid = tid >> 5;
  int b = blockIdx.z, h = blockIdx.y;
  int ig0 = blockIdx.x * AT_R;
  const long cb = (long)b * SEQ * CH + h * HDIM;
  const __bf16* kbase = kc + cb;
  const __bf16* vbase = vc + cb;

  // stage Q block [32 x 128]
#pragma unroll
  for (int p = 0; p < 2; ++p) {
    int idx = p * 256 + tid;
    int r = idx >> 4, ck = idx & 15;
    uint4 v = *reinterpret_cast<const uint4*>(qc + cb + (long)(ig0 + r) * CH + ck * 8);
    *reinterpret_cast<uint4*>(&Qs[r * KVLD + ck * 8]) = v;
  }

  int rr = wid & 1;   // row fragment (16 rows)
  int cp = wid >> 1;  // column pair (2 x 16 cols)

  // ---- phase 1: S = scale * Q Kt  (TDM-streamed K tiles) ----
  if (wid == 0)
    tdm_load_2d(ldsaddr(KV0p), kbase, 128, 128, CH, 5, 3);
  for (int jb = 0; jb < SEQ / 128; ++jb) {
    int buf = jb & 1;
    __bf16* cur = (buf == 0) ? KV0p : KV1p;
    __bf16* nxt = (buf == 0) ? KV1p : KV0p;
    if (wid == 0) {
      if (jb + 1 < SEQ / 128) {
        tdm_load_2d(ldsaddr(nxt), kbase + (long)(jb + 1) * 128 * CH,
                    128, 128, CH, 5, 3);
        __builtin_amdgcn_s_wait_tensorcnt(1);
      } else {
        __builtin_amdgcn_s_wait_tensorcnt(0);
      }
    }
    __syncthreads();
    v8f acc0 = zero8(), acc1 = zero8();
#pragma unroll
    for (int ks = 0; ks < 4; ++ks) {
      v16bf af = load_frag_a(Qs, rr * 16, ks * 32, KVLD, 1);
      v16bf b0 = load_frag_b(cur, cp * 32, ks * 32, KVLD, 1);
      v16bf b1 = load_frag_b(cur, cp * 32 + 16, ks * 32, KVLD, 1);
      acc0 = wmma_bf16(af, b0, acc0);
      acc1 = wmma_bf16(af, b1, acc1);
    }
#pragma unroll
    for (int r = 0; r < 8; ++r) {
      int row = rr * 16 + (lane >> 4) * 8 + r;
      int c0 = jb * 128 + cp * 32 + (lane & 15);
      S[row * SEQ + c0]      = acc0[r] * scale;
      S[row * SEQ + c0 + 16] = acc1[r] * scale;
    }
    __syncthreads();   // tile consumed before TDM refills it
  }

  // ---- phase 1b: add skew term S[i][j] += scale * QEt[i,j] ----
  const __bf16* qe = qet + (((long)b * HEADS + h) * SEQ + ig0) * SEQ;
  for (int idx = tid; idx < AT_R * SEQ; idx += 256)
    S[idx] += scale * (float)qe[idx];
  __syncthreads();

  // ---- phase 2: rowwise softmax (unnormalized P, track rowsum) ----
  for (int row = wid; row < AT_R; row += 8) {
    float m = -3.0e38f;
    for (int j = lane; j < SEQ; j += 32) m = fmaxf(m, S[row * SEQ + j]);
#pragma unroll
    for (int o = 16; o > 0; o >>= 1) m = fmaxf(m, __shfl_xor(m, o, 32));
    float sum = 0.f;
    for (int j = lane; j < SEQ; j += 32) {
      float e = __expf(S[row * SEQ + j] - m);
      P[row * SEQ + j] = (__bf16)e;
      sum += e;
    }
#pragma unroll
    for (int o = 16; o > 0; o >>= 1) sum += __shfl_xor(sum, o, 32);
    if (lane == 0) rowsum[row] = sum;
  }
  __syncthreads();

  // ---- phase 3: O = P V (TDM-streamed V tiles), scaled by 1/rowsum ----
  v8f acc0 = zero8(), acc1 = zero8();
  if (wid == 0)
    tdm_load_2d(ldsaddr(KV0p), vbase, 128, 128, CH, 5, 3);
  for (int kb = 0; kb < SEQ / 128; ++kb) {
    int buf = kb & 1;
    __bf16* cur = (buf == 0) ? KV0p : KV1p;
    __bf16* nxt = (buf == 0) ? KV1p : KV0p;
    if (wid == 0) {
      if (kb + 1 < SEQ / 128) {
        tdm_load_2d(ldsaddr(nxt), vbase + (long)(kb + 1) * 128 * CH,
                    128, 128, CH, 5, 3);
        __builtin_amdgcn_s_wait_tensorcnt(1);
      } else {
        __builtin_amdgcn_s_wait_tensorcnt(0);
      }
    }
    __syncthreads();
#pragma unroll
    for (int ks = 0; ks < 4; ++ks) {
      v16bf af = load_frag_a(P, rr * 16, kb * 128 + ks * 32, SEQ, 1);
      v16bf b0 = load_frag_b(cur, cp * 32, ks * 32, 1, KVLD);   // B[n,k]=V[k,n]
      v16bf b1 = load_frag_b(cur, cp * 32 + 16, ks * 32, 1, KVLD);
      acc0 = wmma_bf16(af, b0, acc0);
      acc1 = wmma_bf16(af, b1, acc1);
    }
    __syncthreads();
  }
#pragma unroll
  for (int r = 0; r < 8; ++r) {
    int row = rr * 16 + (lane >> 4) * 8 + r;
    float inv = 1.0f / rowsum[row];
    int c0 = cp * 32 + (lane & 15);
    long ob = cb + (long)(ig0 + row) * CH;
    oc[ob + c0]      = (__bf16)(acc0[r] * inv);
    oc[ob + c0 + 16] = (__bf16)(acc1[r] * inv);
  }
}

// ------------------------------------------------------------------ launch ---

extern "C" void kernel_launch(void* const* d_in, const int* in_sizes, int n_in,
                              void* d_out, int out_size, void* d_ws, size_t ws_size,
                              hipStream_t stream) {
  (void)in_sizes; (void)n_in; (void)out_size; (void)ws_size;
  const float* x     = (const float*)d_in[0];
  const float* wq    = (const float*)d_in[1];
  const float* wk    = (const float*)d_in[2];
  const float* wv    = (const float*)d_in[3];
  const float* wo    = (const float*)d_in[4];
  const float* qconv = (const float*)d_in[5];
  const float* er    = (const float*)d_in[6];
  float* out = (float*)d_out;

  char* ws = (char*)d_ws;
  size_t off = 0;
  auto alloc = [&](size_t bytes) {
    size_t o = off;
    off += (bytes + 255) & ~(size_t)255;
    return o;
  };
  __bf16* wqkv_b = (__bf16*)(ws + alloc((size_t)3 * CH * CH * 2));   // wq|wk|wv
  __bf16* wo_b   = (__bf16*)(ws + alloc((size_t)CH * CH * 2));
  __bf16* xb     = (__bf16*)(ws + alloc((size_t)MPROJ * CH * 2));
  __bf16* ert    = (__bf16*)(ws + alloc((size_t)SEQ * HDIM * 2));
  __bf16* lin_b  = (__bf16*)(ws + alloc((size_t)3 * MPROJ * CH * 2)); // pre-conv q|k|v
  __bf16* qkv_b  = (__bf16*)(ws + alloc((size_t)3 * MPROJ * CH * 2)); // post-conv q|k|v
  __bf16* qet    = (__bf16*)(ws + alloc((size_t)BATCH * HEADS * SEQ * SEQ * 2));
  __bf16* ocb    = (__bf16*)(ws + alloc((size_t)MPROJ * CH * 2));
  __bf16* qcb = qkv_b;
  __bf16* kcb = qkv_b + (size_t)MPROJ * CH;
  __bf16* vcb = qkv_b + (size_t)2 * MPROJ * CH;

  // 1) convert inputs to bf16 (+ transpose er)
  cvt_f32_bf16<<<(MPROJ * CH) / 256, 256, 0, stream>>>(x, xb, MPROJ * CH);
  cvt_f32_bf16<<<(CH * CH) / 256, 256, 0, stream>>>(wq, wqkv_b, CH * CH);
  cvt_f32_bf16<<<(CH * CH) / 256, 256, 0, stream>>>(wk, wqkv_b + (size_t)CH * CH, CH * CH);
  cvt_f32_bf16<<<(CH * CH) / 256, 256, 0, stream>>>(wv, wqkv_b + (size_t)2 * CH * CH, CH * CH);
  cvt_f32_bf16<<<(CH * CH) / 256, 256, 0, stream>>>(wo, wo_b, CH * CH);
  transpose_er<<<(HDIM * SEQ) / 256, 256, 0, stream>>>(er, ert);

  // 2) q/k/v projections in one batched launch: lin[z] = x @ w[z]^T (bf16 out)
  dim3 gp3(CH / GBN, MPROJ / GBM, 3);
  gemm_nt_kernel<__bf16><<<gp3, 256, 0, stream>>>(
      xb, wqkv_b, lin_b, MPROJ, CH, CH, CH, CH, CH,
      /*aO*/0, /*aI*/0, /*bO*/(long)CH * CH, /*bI*/0,
      /*cO*/(long)MPROJ * CH, /*cI*/0, /*innerN*/1, 1.0f);

  // 3) depthwise conv (applied to q, k and v — faithful to reference)
  dim3 gc((SEQ * CH) / 256, BATCH, 3);
  dwconv_bf16<<<gc, 256, 0, stream>>>(lin_b, qconv, qkv_b);

  // 4) skew term: QEt[b,h,i,j] = sum_d ert[i,d] * q[b,h,j,d]
  dim3 gq(SEQ / GBN, SEQ / GBM, BATCH * HEADS);
  gemm_nt_kernel<__bf16><<<gq, 256, 0, stream>>>(
      ert, qcb, qet, SEQ, SEQ, HDIM, HDIM, CH, SEQ,
      /*aO*/0, /*aI*/0,
      /*bO*/(long)SEQ * CH, /*bI*/(long)HDIM,
      /*cO*/(long)HEADS * SEQ * SEQ, /*cI*/(long)SEQ * SEQ,
      /*innerN*/HEADS, 1.0f);

  // 5) attention (score strip + softmax + PV) with TDM-fed big-LDS tiles
  dim3 ga(SEQ / AT_R, HEADS, BATCH);
  attention_kernel<<<ga, 256, (size_t)SM_TOT, stream>>>(
      qcb, kcb, vcb, qet, ocb, 0.022097086912079608f /* 1/sqrt(2048) */);

  // 6) output projection: out = o @ wo^T (f32)
  dim3 gp(CH / GBN, MPROJ / GBM, 1);
  gemm_nt_kernel<float><<<gp, 256, 0, stream>>>(ocb, wo_b, out, MPROJ, CH, CH,
                                                CH, CH, CH, 0, 0, 0, 0, 0, 0, 1, 1.0f);
}
